// StateDreamerModel_85916525789923
// MI455X (gfx1250) — compile-verified, matrix operand order checked by reading, not run on hardware
//
#include <hip/hip_runtime.h>
#include <hip/hip_bf16.h>

// ---------------- problem constants (match reference) ----------------
#define Bn   16384
#define Hn   15
#define Sn   32
#define Dn   256
#define HIDn 256
#define An   6
#define SDn  288            // S + D
#define WAVES 4             // waves per block; each wave owns 16 batch rows

typedef __attribute__((ext_vector_type(16))) __bf16 v16bf;
typedef __attribute__((ext_vector_type(8)))  float  v8f;

// ---------------- scalar activation helpers ----------------
__device__ __forceinline__ float eluf(float x)      { return x > 0.f ? x : __expf(x) - 1.f; }
__device__ __forceinline__ float softplusf(float x) { return x > 20.f ? x : log1pf(__expf(x)); }
__device__ __forceinline__ float sigmoidf(float x)  { return 1.f / (1.f + __expf(-x)); }

// ---------------- register-resident A-matrix fragments ---------------------
// ISA 7.12.2 16-bit A 16x32 layout: lane<16 -> row=lane, holds K=kb..kb+7
// (elems 0..7) and kb+16..kb+23 (elems 8..15); lanes 16..31 use kb+8.
template <int KC>
struct AReg { v16bf f[KC]; };

template <int KC>
__device__ __forceinline__ void load_a_frags(AReg<KC>& A,
                                             const __bf16* __restrict__ act,
                                             int lda, int lane)
{
    const int row = lane & 15;
    const int kb  = (lane >> 4) << 3;
    const __bf16* ap = act + row * lda + kb;
    #pragma unroll
    for (int k = 0; k < KC; k++) {
        #pragma unroll
        for (int e = 0; e < 8; e++) {
            A.f[k][e]     = ap[k * 32 + e];
            A.f[k][e + 8] = ap[k * 32 + 16 + e];
        }
    }
}

// ---------------- WMMA tile GEMM with A in registers -----------------------
// D(16x16) = A(16xK, regs) * W^T(n0..n0+15 rows of [N][kpad] bf16) + bias
template <int KC>
__device__ __forceinline__ v8f gemm_rowtile(const AReg<KC>& A,
                                            const __bf16* __restrict__ W, int kpad,
                                            int n0,
                                            const float* __restrict__ bias, int nbias,
                                            int lane)
{
    v8f c = {0.f, 0.f, 0.f, 0.f, 0.f, 0.f, 0.f, 0.f};
    const int col = n0 + (lane & 15);
    const int kb  = (lane >> 4) << 3;
    const __bf16* bp = W + (size_t)col * kpad + kb;
    #pragma unroll
    for (int k = 0; k < KC; k++) {
        v16bf b;
        #pragma unroll
        for (int e = 0; e < 8; e++) {
            b[e]     = bp[k * 32 + e];
            b[e + 8] = bp[k * 32 + 16 + e];
        }
        c = __builtin_amdgcn_wmma_f32_16x16x32_bf16(false, A.f[k], false, b,
                                                    (short)0, c, false, false);
    }
    float bv = (col < nbias) ? bias[col] : 0.f;
    #pragma unroll
    for (int v = 0; v < 8; v++) c[v] += bv;
    return c;
}

// store a D tile into a bf16 LDS activation buffer (optional ELU)
__device__ __forceinline__ void store_bf16_tile(__bf16* __restrict__ dst, int ldd,
                                                int n0, v8f c, int lane, int do_elu)
{
    const int col = n0 + (lane & 15);
    const int rb  = (lane >> 4) << 3;
    #pragma unroll
    for (int v = 0; v < 8; v++) {
        float x = c[v];
        if (do_elu) x = eluf(x);
        dst[(rb + v) * ldd + col] = (__bf16)x;
    }
}

// ---------------- weight fp32 -> padded bf16 conversion --------------------
__global__ void cvt_pad_kernel(const float* __restrict__ src, __bf16* __restrict__ dst,
                               int N, int K, int Kpad, int total)
{
    int i = blockIdx.x * blockDim.x + threadIdx.x;
    if (i >= total) return;
    int n = i / Kpad;
    int k = i - n * Kpad;
    float v = (n < N && k < K) ? src[n * K + k] : 0.f;
    dst[i] = (__bf16)v;
}

// ---------------- main dreamer rollout kernel ------------------------------
__global__ void __launch_bounds__(32 * WAVES)
dream_kernel(const float* __restrict__ stoch0, const float* __restrict__ gru0,
             const float* __restrict__ anoise, const float* __restrict__ snoise,
             const __bf16* __restrict__ Wa0, const float* __restrict__ ba0,
             const __bf16* __restrict__ Wa1, const float* __restrict__ ba1,
             const __bf16* __restrict__ Wam, const float* __restrict__ bam,
             const __bf16* __restrict__ Was, const float* __restrict__ bas,
             const __bf16* __restrict__ Wimg, const float* __restrict__ bimg,
             const __bf16* __restrict__ Wih, const float* __restrict__ bih,
             const __bf16* __restrict__ Whh, const float* __restrict__ bhh,
             const __bf16* __restrict__ Wp0, const float* __restrict__ bp0,
             const __bf16* __restrict__ Wpm, const float* __restrict__ bpm,
             const __bf16* __restrict__ Wps, const float* __restrict__ bps,
             float* __restrict__ out)
{
    extern __shared__ char smem[];
    const int lane = threadIdx.x & 31;
    const int wave = threadIdx.x >> 5;

    // per-wave LDS slices (wave-private => no barriers anywhere)
    __bf16* stA = (__bf16*)smem + (size_t)wave * (16 * SDn + 16 * HIDn + 16 * HIDn);
    __bf16* stB = stA + 16 * SDn;    // 16 x 256
    __bf16* stC = stB + 16 * HIDn;   // 16 x 256

    const int rowBase = blockIdx.x * (WAVES * 16) + wave * 16;
    const int col = lane & 15;
    const int rb  = (lane >> 4) << 3;

    // stage initial state: stA[r][0..31] = stoch, stA[r][32..287] = gru
    for (int i = lane; i < 16 * SDn; i += 32) {
        int r = i / SDn, cc = i % SDn;
        float v = (cc < Sn) ? stoch0[(size_t)(rowBase + r) * Sn + cc]
                            : gru0[(size_t)(rowBase + r) * Dn + (cc - Sn)];
        stA[r * SDn + cc] = (__bf16)v;
    }

    for (int t = 0; t < Hn; t++) {
        float* outt = out + (size_t)t * Bn * SDn;

        // ---- action MLP layer 0: ah0 = elu(state @ Wa0^T + b), K=288 -> stB
        {
            AReg<9> As;
            load_a_frags(As, stA, SDn, lane);
            #pragma unroll 1
            for (int nt = 0; nt < HIDn / 16; nt++) {
                v8f c = gemm_rowtile(As, Wa0, SDn, nt * 16, ba0, HIDn, lane);
                store_bf16_tile(stB, HIDn, nt * 16, c, lane, 1);
            }
        }
        // ---- layer 1: ah1 = elu(ah0 @ Wa1^T + b) -> stC
        {
            AReg<8> As;
            load_a_frags(As, stB, HIDn, lane);
            #pragma unroll 1
            for (int nt = 0; nt < HIDn / 16; nt++) {
                v8f c = gemm_rowtile(As, Wa1, HIDn, nt * 16, ba1, HIDn, lane);
                store_bf16_tile(stC, HIDn, nt * 16, c, lane, 1);
            }
        }
        // ---- build img input in stB[r][0..63]: [stoch | action | zero-pad]
        for (int i = lane; i < 16 * 64; i += 32) {
            int r = i >> 6, cc = i & 63;
            stB[r * HIDn + cc] = (cc < Sn) ? stA[r * SDn + cc] : (__bf16)0.f;
        }
        // ---- action heads (N=6 padded to 16): action = mean + softplus(std)*noise
        {
            AReg<8> As;
            load_a_frags(As, stC, HIDn, lane);
            v8f cm = gemm_rowtile(As, Wam, HIDn, 0, bam, An, lane);
            v8f cs = gemm_rowtile(As, Was, HIDn, 0, bas, An, lane);
            if (col < An) {
                #pragma unroll
                for (int v = 0; v < 8; v++) {
                    int rg = rowBase + rb + v;
                    float nz = anoise[(size_t)t * Bn * An + (size_t)rg * An + col];
                    float av = cm[v] + softplusf(cs[v]) * nz;
                    stB[(rb + v) * HIDn + Sn + col] = (__bf16)av;
                }
            }
        }
        // ---- img layer: x = elu(imgin @ Wimg^T + b), K=64 (zero-padded) -> stC
        {
            AReg<2> As;
            load_a_frags(As, stB, HIDn, lane);
            #pragma unroll 1
            for (int nt = 0; nt < HIDn / 16; nt++) {
                v8f c = gemm_rowtile(As, Wimg, 64, nt * 16, bimg, HIDn, lane);
                store_bf16_tile(stC, HIDn, nt * 16, c, lane, 1);
            }
        }
        // ---- GRU gates, fused per 16-col tile (r/z/n share lane mapping).
        //      x in stC, old h in stA[.,32..287]; new h -> stB + fp32 out.
        //      Both A operand sets live in registers across all 48 tile-GEMMs.
        {
            AReg<8> Ax, Ah;
            load_a_frags(Ax, stC, HIDn, lane);
            load_a_frags(Ah, stA + Sn, SDn, lane);
            #pragma unroll 1
            for (int ct = 0; ct < Dn / 16; ct++) {
                int c0 = ct * 16;
                v8f gir = gemm_rowtile(Ax, Wih, HIDn, c0,          bih, 3 * Dn, lane);
                v8f ghr = gemm_rowtile(Ah, Whh, HIDn, c0,          bhh, 3 * Dn, lane);
                v8f giz = gemm_rowtile(Ax, Wih, HIDn, Dn + c0,     bih, 3 * Dn, lane);
                v8f ghz = gemm_rowtile(Ah, Whh, HIDn, Dn + c0,     bhh, 3 * Dn, lane);
                v8f gin = gemm_rowtile(Ax, Wih, HIDn, 2 * Dn + c0, bih, 3 * Dn, lane);
                v8f ghn = gemm_rowtile(Ah, Whh, HIDn, 2 * Dn + c0, bhh, 3 * Dn, lane);
                #pragma unroll
                for (int v = 0; v < 8; v++) {
                    int r = rb + v;
                    float rg = sigmoidf(gir[v] + ghr[v]);
                    float zg = sigmoidf(giz[v] + ghz[v]);
                    float ng = tanhf(gin[v] + rg * ghn[v]);
                    float hold = (float)stA[r * SDn + Sn + c0 + col];
                    float hnew = (1.f - zg) * ng + zg * hold;
                    stB[r * HIDn + c0 + col] = (__bf16)hnew;
                    outt[(size_t)(rowBase + r) * SDn + Sn + c0 + col] = hnew;
                }
            }
        }
        // ---- prior MLP: ph = elu(hnew @ Wp0^T + b) -> stC
        {
            AReg<8> As;
            load_a_frags(As, stB, HIDn, lane);
            #pragma unroll 1
            for (int nt = 0; nt < HIDn / 16; nt++) {
                v8f c = gemm_rowtile(As, Wp0, Dn, nt * 16, bp0, HIDn, lane);
                store_bf16_tile(stC, HIDn, nt * 16, c, lane, 1);
            }
        }
        // ---- prior heads: stoch' = mean + softplus(std)*noise -> stA[.,0..31] + out
        {
            AReg<8> As;
            load_a_frags(As, stC, HIDn, lane);
            #pragma unroll 1
            for (int s2 = 0; s2 < Sn / 16; s2++) {
                int n0 = s2 * 16;
                v8f cm = gemm_rowtile(As, Wpm, HIDn, n0, bpm, Sn, lane);
                v8f cs = gemm_rowtile(As, Wps, HIDn, n0, bps, Sn, lane);
                #pragma unroll
                for (int v = 0; v < 8; v++) {
                    int r  = rb + v;
                    int rg = rowBase + r;
                    float nz = snoise[(size_t)t * Bn * Sn + (size_t)rg * Sn + n0 + col];
                    float sv = cm[v] + softplusf(cs[v]) * nz;
                    stA[r * SDn + n0 + col] = (__bf16)sv;
                    outt[(size_t)rg * SDn + n0 + col] = sv;
                }
            }
        }
        // ---- commit new h into the state buffer for the next step
        for (int i = lane; i < 16 * Dn; i += 32) {
            int r = i >> 8, cc = i & 255;
            stA[r * SDn + Sn + cc] = stB[r * HIDn + cc];
        }
    }
}

// ---------------- host-side launch ----------------------------------------
extern "C" void kernel_launch(void* const* d_in, const int* in_sizes, int n_in,
                              void* d_out, int out_size, void* d_ws, size_t ws_size,
                              hipStream_t stream)
{
    (void)in_sizes; (void)n_in; (void)out_size; (void)ws_size;
    const float* stoch0 = (const float*)d_in[0];
    const float* gru0   = (const float*)d_in[1];
    const float* anoise = (const float*)d_in[2];
    const float* snoise = (const float*)d_in[3];
    const float* ba0 = (const float*)d_in[5];
    const float* ba1 = (const float*)d_in[7];
    const float* bam = (const float*)d_in[9];
    const float* bas = (const float*)d_in[11];
    const float* bimg = (const float*)d_in[13];
    const float* bih = (const float*)d_in[15];
    const float* bhh = (const float*)d_in[17];
    const float* bp0 = (const float*)d_in[19];
    const float* bpm = (const float*)d_in[21];
    const float* bps = (const float*)d_in[23];

    __bf16* ws = (__bf16*)d_ws;
    size_t off = 0;
    auto cvt = [&](int idx, int N, int K, int Kpad, int Npad) -> __bf16* {
        __bf16* dst = ws + off;
        int total = Npad * Kpad;
        off += (size_t)total;
        cvt_pad_kernel<<<(total + 255) / 256, 256, 0, stream>>>(
            (const float*)d_in[idx], dst, N, K, Kpad, total);
        return dst;
    };
    __bf16* Wa0  = cvt(4,  HIDn,   SDn, SDn, HIDn);     // 256 x 288
    __bf16* Wa1  = cvt(6,  HIDn,  HIDn, HIDn, HIDn);    // 256 x 256
    __bf16* Wam  = cvt(8,  An,    HIDn, HIDn, 16);      // 6->16 x 256
    __bf16* Was  = cvt(10, An,    HIDn, HIDn, 16);
    __bf16* Wimg = cvt(12, HIDn,  Sn + An, 64, HIDn);   // 256 x 38->64
    __bf16* Wih  = cvt(14, 3 * Dn, HIDn, HIDn, 3 * Dn); // 768 x 256
    __bf16* Whh  = cvt(16, 3 * Dn, Dn,   Dn,   3 * Dn); // 768 x 256
    __bf16* Wp0  = cvt(18, HIDn,  Dn,   Dn,   HIDn);    // 256 x 256
    __bf16* Wpm  = cvt(20, Sn,    HIDn, HIDn, Sn);      // 32 x 256
    __bf16* Wps  = cvt(22, Sn,    HIDn, HIDn, Sn);

    const size_t lds_bytes =
        (size_t)WAVES * (16 * SDn + 16 * HIDn + 16 * HIDn) * sizeof(__bf16); // 102400 B

    dream_kernel<<<Bn / (WAVES * 16), 32 * WAVES, lds_bytes, stream>>>(
        stoch0, gru0, anoise, snoise,
        Wa0, ba0, Wa1, ba1, Wam, bam, Was, bas, Wimg, bimg,
        Wih, bih, Whh, bhh, Wp0, bp0, Wpm, bpm, Wps, bps,
        (float*)d_out);
}
